// STULayer_30562987278836
// MI455X (gfx1250) — matrix operand order repaired and checked by reading, not compiled, not run on hardware
//
#include <hip/hip_runtime.h>
#include <hip/hip_bf16.h>

// Problem constants (match reference)
#define SEQ  2048
#define KF   16
#define DIN  512
#define DOUT 512
#define BB   2

typedef __attribute__((ext_vector_type(16))) _Float16 v16h;
typedef __attribute__((ext_vector_type(8)))  _Float16 v8h;
typedef __attribute__((ext_vector_type(8)))  float    v8f;

static __device__ __forceinline__ v16h cat8(v8h lo, v8h hi) {
  return __builtin_shufflevector(lo, hi, 0,1,2,3,4,5,6,7,8,9,10,11,12,13,14,15);
}

static __device__ __forceinline__ v8f wmma_f16(v16h a, v16h b, v8f c) {
  // D = A(16x32 f16) * B(32x16 f16) + C(16x16 f32)
  return __builtin_amdgcn_wmma_f32_16x16x32_f16(false, a, false, b, (short)0, c,
                                                false, false);
}

// ---------------- pack kernels ----------------

// xp = f16(x), xm = f16(x * sgn(t)), sgn(t) = (-1)^t
__global__ void pack_x_kernel(const float* __restrict__ x,
                              _Float16* __restrict__ xp,
                              _Float16* __restrict__ xm) {
  size_t i = (size_t)blockIdx.x * blockDim.x + threadIdx.x;
  float v = x[i];
  int t = (int)((i / DIN) % SEQ);
  xp[i] = (_Float16)v;
  xm[i] = (_Float16)((t & 1) ? -v : v);
}

__global__ void zero_out_kernel(float* __restrict__ out) {
  size_t i = (size_t)blockIdx.x * blockDim.x + threadIdx.x;
  out[i] = 0.0f;
}

// Transpose M[k][i][o] (f32) -> Mt[k][o][i] (f16), LDS-tiled for coalescing.
__global__ void pack_MT_kernel(const float* __restrict__ Mp,
                               const float* __restrict__ Mm,
                               _Float16* __restrict__ Mtp,
                               _Float16* __restrict__ Mtm) {
  __shared__ float tile[32][33];
  const float* src = blockIdx.z ? Mm : Mp;
  _Float16*    dst = blockIdx.z ? Mtm : Mtp;
  int k  = blockIdx.y;
  int it = (blockIdx.x & (DIN / 32 - 1)) * 32;           // i tile
  int ot = (blockIdx.x / (DIN / 32)) * 32;               // o tile
  int tx = threadIdx.x, ty = threadIdx.y;
  const float* s = src + (size_t)k * DIN * DOUT;
#pragma unroll
  for (int r = 0; r < 4; ++r)
    tile[ty + 8 * r][tx] = s[(size_t)(it + ty + 8 * r) * DOUT + (ot + tx)];
  __syncthreads();
  _Float16* d = dst + (size_t)k * DOUT * DIN;
#pragma unroll
  for (int r = 0; r < 4; ++r)
    d[(size_t)(ot + ty + 8 * r) * DIN + (it + tx)] = (_Float16)tile[tx][ty + 8 * r];
}

// Toeplitz materialization: T[l][t] = phi[l-t, k] for t<=l else 0  (f16, row-major)
__global__ void pack_T_kernel(const float* __restrict__ phi,
                              _Float16* __restrict__ T, int k) {
  size_t i = (size_t)blockIdx.x * blockDim.x + threadIdx.x;
  int l = (int)(i / SEQ), t = (int)(i % SEQ);
  float v = (t <= l) ? phi[(size_t)(l - t) * KF + k] : 0.0f;
  T[i] = (_Float16)v;
}

// ---------------- stage 1: Y_k = x @ M_k, stored transposed Y[b][o][t] ----------------
// WG tile: 128(t) x 128(o); 8 waves = 2 row-groups x 4 col-groups.
// Per wave: 4 row-frags x 2 col-frags = 8 WMMAs / K-step.
// grid: x = (t_tile*4 + o_tile) [16*4=64], y = b, z = sign; block = 256
__global__ void stage1_kernel(const _Float16* __restrict__ xp,
                              const _Float16* __restrict__ xm,
                              const _Float16* __restrict__ Mtp_k,
                              const _Float16* __restrict__ Mtm_k,
                              _Float16* __restrict__ Ytp,
                              _Float16* __restrict__ Ytm) {
  int b = blockIdx.y, s = blockIdx.z;
  int ot = (blockIdx.x & 3) * 128;
  int tt = (blockIdx.x >> 2) * 128;
  const _Float16* X = (s ? xm : xp) + (size_t)b * SEQ * DIN;    // [t][i]
  const _Float16* M = s ? Mtm_k : Mtp_k;                        // [o][i]
  _Float16*       Y = (s ? Ytm : Ytp) + (size_t)b * DOUT * SEQ; // [o][t]

  int w = threadIdx.x >> 5, ln = threadIdx.x & 31;
  int wr = w & 1, wc = w >> 1;       // 2 t-halves x 4 o-groups of 32
  int m = ln & 15, hi = ln >> 4;
  int ow = ot + 32 * wc;

  v8f c[4][2] = {};
  for (int i0 = 0; i0 < DIN; i0 += 32) {
    v16h Bf[2];
#pragma unroll
    for (int cb = 0; cb < 2; ++cb) {
      const _Float16* bp = M + (size_t)(ow + 16 * cb + m) * DIN + i0 + 16 * hi;
      Bf[cb] = cat8(*(const v8h*)bp, *(const v8h*)(bp + 8));
    }
#pragma unroll
    for (int rr = 0; rr < 4; ++rr) {
      int trow = tt + 64 * wr + 16 * rr + m;
      const _Float16* ar = X + (size_t)trow * DIN + i0 + 16 * hi;
      v16h a = cat8(*(const v8h*)ar, *(const v8h*)(ar + 8));
#pragma unroll
      for (int cb = 0; cb < 2; ++cb)
        c[rr][cb] = wmma_f16(a, Bf[cb], c[rr][cb]);
    }
  }
  // C layout: lane owns column n=m, 8 consecutive rows -> contiguous along t
#pragma unroll
  for (int rr = 0; rr < 4; ++rr) {
    int tbase = tt + 64 * wr + 16 * rr + 8 * hi;
#pragma unroll
    for (int cb = 0; cb < 2; ++cb) {
      v8h r0;
#pragma unroll
      for (int r = 0; r < 8; ++r) r0[r] = (_Float16)c[rr][cb][r];
      *(v8h*)(Y + (size_t)(ow + 16 * cb + m) * SEQ + tbase) = r0;
    }
  }
}

// ---------------- stage 2: out[b,l,o] += sum_{t<=l} T[l][t] * Y[b][o][t] ----------------
// WG tile: 128(l) x 128(o); A (T rows) staged per K-step into LDS via async-to-LDS,
// shared by all 8 waves. Per wave: 4 row x 2 col x 2 signs = 16 WMMAs / step.
// grid: x = (l_tile*4 + o_tile) [16*4=64], y = b
#define A_PITCH 40  // halves; 80B rows: 16B-aligned, breaks 16-bank stride
__global__ void conv_kernel(const _Float16* __restrict__ T,
                            const _Float16* __restrict__ Ytp,
                            const _Float16* __restrict__ Ytm,
                            float* __restrict__ out) {
  __shared__ _Float16 As[128 * A_PITCH];
  int b = blockIdx.y;
  int ot = (blockIdx.x & 3) * 128;
  int l0 = (blockIdx.x >> 2) * 128;
  const _Float16* Yp = Ytp + (size_t)b * DOUT * SEQ;
  const _Float16* Ym = Ytm + (size_t)b * DOUT * SEQ;

  int w = threadIdx.x >> 5, ln = threadIdx.x & 31;
  int wr = w & 1, wc = w >> 1;
  int m = ln & 15, hi = ln >> 4;
  int ow = ot + 32 * wc;

  v8f cp[4][2] = {}, cm[4][2] = {};
  uint32_t lds_base = (uint32_t)(uintptr_t)(&As[0]);

  const int tend = l0 + 128;  // causality: T is zero above the diagonal
  for (int t0 = 0; t0 < tend; t0 += 32) {
    __syncthreads();  // previous step's A reads complete before overwrite
    // Stage A tile T[l0..l0+128][t0..t0+32] (8KB) via async DMA to LDS:
    // 512 x 16B chunks, 2 per thread.
#pragma unroll
    for (int c = 0; c < 2; ++c) {
      int ch  = (int)threadIdx.x + 256 * c;    // 0..511
      int row = ch >> 2;                       // 0..127
      int col = (ch & 3) * 8;                  // 0,8,16,24 (halves)
      unsigned long long ga =
          (unsigned long long)(uintptr_t)(T + (size_t)(l0 + row) * SEQ + t0 + col);
      uint32_t la = lds_base + (uint32_t)(row * A_PITCH + col) * 2u;
      asm volatile("global_load_async_to_lds_b128 %0, %1, off"
                   :: "v"(la), "v"(ga) : "memory");
    }
    asm volatile("s_wait_asynccnt 0x0" ::: "memory");
    __syncthreads();

    // B fragments from L2-resident Y (direct b128 streams)
    v16h Bp[2], Bm[2];
#pragma unroll
    for (int cb = 0; cb < 2; ++cb) {
      const _Float16* bp = Yp + (size_t)(ow + 16 * cb + m) * SEQ + t0 + 16 * hi;
      const _Float16* bm = Ym + (size_t)(ow + 16 * cb + m) * SEQ + t0 + 16 * hi;
      Bp[cb] = cat8(*(const v8h*)bp, *(const v8h*)(bp + 8));
      Bm[cb] = cat8(*(const v8h*)bm, *(const v8h*)(bm + 8));
    }
    // A fragments from LDS (ds_load_b128 x2 per frag), reused across signs+cols
#pragma unroll
    for (int rr = 0; rr < 4; ++rr) {
      const _Float16* ar = As + (64 * wr + 16 * rr + m) * A_PITCH + 16 * hi;
      v16h a = cat8(*(const v8h*)ar, *(const v8h*)(ar + 8));
#pragma unroll
      for (int cb = 0; cb < 2; ++cb) {
        cp[rr][cb] = wmma_f16(a, Bp[cb], cp[rr][cb]);
        cm[rr][cb] = wmma_f16(a, Bm[cb], cm[rr][cb]);
      }
    }
  }

  // epilogue: out += Cplus + sgn(l) * Cminus   (RMW; accumulates across k launches)
  float* ob = out + (size_t)b * SEQ * DOUT;
#pragma unroll
  for (int rr = 0; rr < 4; ++rr) {
    int lbase = l0 + 64 * wr + 16 * rr + 8 * hi;
#pragma unroll
    for (int cb = 0; cb < 2; ++cb) {
      int o = ow + 16 * cb + m;
#pragma unroll
      for (int r = 0; r < 8; ++r) {
        int l = lbase + r;
        float sg = (l & 1) ? -1.0f : 1.0f;
        float* p = ob + (size_t)l * DOUT + o;
        *p += cp[rr][cb][r] + sg * cm[rr][cb][r];
      }
    }
  }
}

// ---------------- launch ----------------
extern "C" void kernel_launch(void* const* d_in, const int* in_sizes, int n_in,
                              void* d_out, int out_size, void* d_ws, size_t ws_size,
                              hipStream_t stream) {
  const float* x   = (const float*)d_in[0];  // [B, L, D_in]
  const float* phi = (const float*)d_in[1];  // [L, K]
  const float* Mp  = (const float*)d_in[2];  // [K, D_in, D_out]
  const float* Mm  = (const float*)d_in[3];  // [K, D_in, D_out]
  float* out = (float*)d_out;                // [B, L, D_out]

  // Workspace layout (f16), ~40 MB total
  _Float16* xp  = (_Float16*)d_ws;                   // [B][L][Din]
  _Float16* xm  = xp  + (size_t)BB * SEQ * DIN;      // [B][L][Din]
  _Float16* Mtp = xm  + (size_t)BB * SEQ * DIN;      // [K][Dout][Din]
  _Float16* Mtm = Mtp + (size_t)KF * DOUT * DIN;     // [K][Dout][Din]
  _Float16* T   = Mtm + (size_t)KF * DOUT * DIN;     // [L][L], per-k reuse
  _Float16* Ytp = T   + (size_t)SEQ * SEQ;           // [B][Dout][L], per-k reuse
  _Float16* Ytm = Ytp + (size_t)BB * DOUT * SEQ;     // [B][Dout][L], per-k reuse

  pack_x_kernel<<<(BB * SEQ * DIN) / 256, 256, 0, stream>>>(x, xp, xm);
  pack_MT_kernel<<<dim3((DIN / 32) * (DOUT / 32), KF, 2), dim3(32, 8), 0, stream>>>(
      Mp, Mm, Mtp, Mtm);
  zero_out_kernel<<<(BB * SEQ * DOUT) / 256, 256, 0, stream>>>(out);

  for (int k = 0; k < KF; ++k) {
    pack_T_kernel<<<((size_t)SEQ * SEQ) / 256, 256, 0, stream>>>(phi, T, k);
    stage1_kernel<<<dim3(64, BB, 2), 256, 0, stream>>>(
        xp, xm, Mtp + (size_t)k * DOUT * DIN, Mtm + (size_t)k * DOUT * DIN, Ytp, Ytm);
    conv_kernel<<<dim3(64, BB), 256, 0, stream>>>(T, Ytp, Ytm, out);
  }
}
